// StructureEncoder_9740985827767
// MI455X (gfx1250) — matrix operand order repaired and checked by reading, not compiled, not run on hardware
//
#include <hip/hip_runtime.h>
#include <hip/hip_bf16.h>

// ---------------- problem constants ----------------
#define NN   20000
#define EE   320000
#define BB   64
#define DM   128
#define RBFD 50
#define GRAMD 9
#define LNUM 4
#define EIN  315
#define EINP 320          // padded K for edge matmul (10 chunks of 32)

// packed-weight layer strides (in ushorts): kTiles * 8 nTiles * 32 lanes * 16 elems
#define EW1P_L (10*8*32*16)
#define EW2P_L (4*8*32*16)
#define NW1P_L (8*8*32*16)
#define NW2P_L (4*8*32*16)

typedef __attribute__((ext_vector_type(16))) __bf16 v16bf;
typedef __attribute__((ext_vector_type(8)))  __bf16 v8bf;
typedef __attribute__((ext_vector_type(8)))  float  v8f;

// float -> bf16 bits, round-to-nearest-even
__device__ __forceinline__ unsigned short f2bf(float f) {
    unsigned u = __float_as_uint(f);
    unsigned r = u + 0x7FFFu + ((u >> 16) & 1u);
    return (unsigned short)(r >> 16);
}

__device__ __forceinline__ float silu_f(float x) {
    return x / (1.0f + __expf(-x));
}

__device__ __forceinline__ float gelu_f(float x) {
    const float c = 0.7978845608028654f;
    float t = tanhf(c * (x + 0.044715f * x * x * x));
    return 0.5f * x * (1.0f + t);
}

// A fragment (16x32 bf16): M across lanes 0..15; per lane-half hf:
// elements 0..7  -> K = hf*8 + 0..7 ; elements 8..15 -> K = 16 + hf*8 + 0..7
__device__ __forceinline__ v16bf load_afrag(const unsigned short* rowk, int hf) {
    v8bf lo = *(const v8bf*)(rowk + hf * 8);
    v8bf hi = *(const v8bf*)(rowk + 16 + hf * 8);
    v16bf a;
#pragma unroll
    for (int i = 0; i < 8; ++i) { a[i] = lo[i]; a[8 + i] = hi[i]; }
    return a;
}

// ---------------- weight packing ----------------
// Pack W[l][k][n] (n-fast, 128 wide) into WMMA B-fragment order, bf16.
// B layout (32x16): per lane-half hf, elems 0..7 -> K = hf*8 + 0..7, elems 8..15 -> K = 16 + hf*8 + 0..7,
// n = nt*16 + lane%16; VGPR j elem t -> k = half*8 + 2j + t + (j >= 4 ? 8 : 0).
__global__ void pack_w_kernel(const float* __restrict__ W,
                              unsigned short* __restrict__ P,
                              int Ksrc, int Kpad, int nLayers) {
    int kTiles = Kpad / 32;
    int total = nLayers * kTiles * 8 * 32;
    int t = blockIdx.x * blockDim.x + threadIdx.x;
    if (t >= total) return;
    int lane = t & 31;
    int nt = (t >> 5) & 7;
    int kt = (t >> 8) % kTiles;
    int l  = (t >> 8) / kTiles;
    int n  = nt * 16 + (lane & 15);
    int hf = lane >> 4;
    unsigned short* dst = P + (size_t)t * 16;
#pragma unroll
    for (int j = 0; j < 8; ++j) {
#pragma unroll
        for (int b = 0; b < 2; ++b) {
            int k = kt * 32 + hf * 8 + 2 * j + b + (j >= 4 ? 8 : 0);
            float v = (k < Ksrc) ? W[((size_t)l * Ksrc + k) * DM + n] : 0.0f;
            dst[2 * j + b] = f2bf(v);
        }
    }
}

// ---------------- small utility kernels ----------------
__global__ void zero_kernel(float* p, size_t n) {
    size_t i = (size_t)blockIdx.x * blockDim.x + threadIdx.x;
    if (i < n) p[i] = 0.0f;
}

__global__ void embed_kernel(const int* __restrict__ at,
                             const float* __restrict__ emb,
                             float* __restrict__ h) {
    int i = blockIdx.x, d = threadIdx.x;
    h[(size_t)i * DM + d] = emb[(size_t)at[i] * DM + d];
}

__global__ void degree_kernel(const int* __restrict__ dstIdx, float* __restrict__ cnt) {
    int e = blockIdx.x * blockDim.x + threadIdx.x;
    if (e < EE) atomicAdd(&cnt[dstIdx[e]], 1.0f);
}

// wave-per-node layernorm (wave32): lane holds 4 of 128 elements
__global__ void ln_kernel(const float* __restrict__ h, float* __restrict__ hn,
                          const float* __restrict__ g, const float* __restrict__ b) {
    int wave = threadIdx.x >> 5;
    int lane = threadIdx.x & 31;
    int node = blockIdx.x * 8 + wave;
    const float4 x4 = *(const float4*)(h + (size_t)node * DM + lane * 4);
    float s = x4.x + x4.y + x4.z + x4.w;
#pragma unroll
    for (int m = 16; m > 0; m >>= 1) s += __shfl_xor(s, m, 32);
    float mu = s * (1.0f / 128.0f);
    float d0 = x4.x - mu, d1 = x4.y - mu, d2 = x4.z - mu, d3 = x4.w - mu;
    float v = d0 * d0 + d1 * d1 + d2 * d2 + d3 * d3;
#pragma unroll
    for (int m = 16; m > 0; m >>= 1) v += __shfl_xor(v, m, 32);
    float inv = rsqrtf(v * (1.0f / 128.0f) + 1e-5f);
    float4 o;
    o.x = d0 * inv * g[lane * 4 + 0] + b[lane * 4 + 0];
    o.y = d1 * inv * g[lane * 4 + 1] + b[lane * 4 + 1];
    o.z = d2 * inv * g[lane * 4 + 2] + b[lane * 4 + 2];
    o.w = d3 * inv * g[lane * 4 + 3] + b[lane * 4 + 3];
    *(float4*)(hn + (size_t)node * DM + lane * 4) = o;
}

// ---------------- edge MLP: one wave per 32-edge tile ----------------
__global__ void __launch_bounds__(32)
edge_kernel(const float* __restrict__ hn,
            const int* __restrict__ srcIdx, const int* __restrict__ dstIdx,
            const float* __restrict__ gram, const float* __restrict__ rbf,
            const unsigned short* __restrict__ w1p, const float* __restrict__ b1,
            const unsigned short* __restrict__ w2p, const float* __restrict__ b2,
            float* __restrict__ msum) {
    __shared__ unsigned short At[32][EINP];  // 32 x 320 bf16 (20 KB)
    __shared__ unsigned short Mt[32][DM];    // 32 x 128 bf16 (8 KB)
    int lane = threadIdx.x;
    int eBase = blockIdx.x * 32;
    const float4* hn4 = (const float4*)hn;

    // coalesced gather: per edge row, the wave fetches the full 512B hn row
#pragma unroll 4
    for (int r = 0; r < 32; ++r) {
        int e = eBase + r;
        float4 v = hn4[(size_t)srcIdx[e] * 32 + lane];
        ushort4 pv; pv.x = f2bf(v.x); pv.y = f2bf(v.y); pv.z = f2bf(v.z); pv.w = f2bf(v.w);
        *(ushort4*)&At[r][lane * 4] = pv;
        float4 u = hn4[(size_t)dstIdx[e] * 32 + lane];
        ushort4 pu; pu.x = f2bf(u.x); pu.y = f2bf(u.y); pu.z = f2bf(u.z); pu.w = f2bf(u.w);
        *(ushort4*)&At[r][128 + lane * 4] = pu;
    }
    {   // gram/rbf: each lane owns its own row (contiguous per-lane stream)
        int e = eBase + lane;
#pragma unroll
        for (int j = 0; j < GRAMD; ++j)
            At[lane][256 + j] = f2bf(gram[(size_t)e * GRAMD + j]);
#pragma unroll
        for (int j = 0; j < RBFD; ++j)
            At[lane][265 + j] = f2bf(rbf[(size_t)e * RBFD + j]);
#pragma unroll
        for (int j = EIN; j < EINP; ++j) At[lane][j] = 0;
    }
    __syncthreads();

    int m = lane & 15, hf = lane >> 4, col = m;

    v8f acc[2][8];
#pragma unroll
    for (int g = 0; g < 2; ++g)
#pragma unroll
        for (int nt = 0; nt < 8; ++nt)
#pragma unroll
            for (int i = 0; i < 8; ++i) acc[g][nt][i] = 0.0f;

#pragma unroll
    for (int kt = 0; kt < 10; ++kt) {
        v16bf a0 = load_afrag(&At[m][kt * 32], hf);
        v16bf a1 = load_afrag(&At[16 + m][kt * 32], hf);
#pragma unroll
        for (int nt = 0; nt < 8; ++nt) {
            v16bf bm = *(const v16bf*)(w1p + (((size_t)kt * 8 + nt) * 32 + lane) * 16);
            acc[0][nt] = __builtin_amdgcn_wmma_f32_16x16x32_bf16(
                false, a0, false, bm, (short)0, acc[0][nt], false, false);
            acc[1][nt] = __builtin_amdgcn_wmma_f32_16x16x32_bf16(
                false, a1, false, bm, (short)0, acc[1][nt], false, false);
        }
    }

#pragma unroll
    for (int nt = 0; nt < 8; ++nt) {
        float bias = b1[nt * 16 + col];
#pragma unroll
        for (int g = 0; g < 2; ++g)
#pragma unroll
            for (int r = 0; r < 8; ++r) {
                float x = acc[g][nt][r] + bias;
                Mt[g * 16 + r + 8 * hf][nt * 16 + col] = f2bf(silu_f(x));
            }
    }
    __syncthreads();

    v8f acc2[2][8];
#pragma unroll
    for (int g = 0; g < 2; ++g)
#pragma unroll
        for (int nt = 0; nt < 8; ++nt)
#pragma unroll
            for (int i = 0; i < 8; ++i) acc2[g][nt][i] = 0.0f;

#pragma unroll
    for (int kt = 0; kt < 4; ++kt) {
        v16bf a0 = load_afrag(&Mt[m][kt * 32], hf);
        v16bf a1 = load_afrag(&Mt[16 + m][kt * 32], hf);
#pragma unroll
        for (int nt = 0; nt < 8; ++nt) {
            v16bf bm = *(const v16bf*)(w2p + (((size_t)kt * 8 + nt) * 32 + lane) * 16);
            acc2[0][nt] = __builtin_amdgcn_wmma_f32_16x16x32_bf16(
                false, a0, false, bm, (short)0, acc2[0][nt], false, false);
            acc2[1][nt] = __builtin_amdgcn_wmma_f32_16x16x32_bf16(
                false, a1, false, bm, (short)0, acc2[1][nt], false, false);
        }
    }

#pragma unroll
    for (int nt = 0; nt < 8; ++nt) {
        float bias = b2[nt * 16 + col];
#pragma unroll
        for (int g = 0; g < 2; ++g)
#pragma unroll
            for (int r = 0; r < 8; ++r) {
                float s = silu_f(acc2[g][nt][r] + bias);
                int e = eBase + g * 16 + r + 8 * hf;
                atomicAdd(&msum[(size_t)dstIdx[e] * DM + nt * 16 + col], s);
            }
    }
}

// ---------------- node MLP: one wave per 32-node tile ----------------
__global__ void __launch_bounds__(32)
node_kernel(const float* __restrict__ hn, const float* __restrict__ msum,
            const float* __restrict__ cnt,
            const unsigned short* __restrict__ w1p, const float* __restrict__ b1,
            const unsigned short* __restrict__ w2p, const float* __restrict__ b2,
            float* __restrict__ h) {
    __shared__ unsigned short At[32][256];   // 16 KB
    __shared__ unsigned short Mt[32][DM];    // 8 KB
    int lane = threadIdx.x;
    int nBase = blockIdx.x * 32;
    const float4* hn4 = (const float4*)hn;
    const float4* ms4 = (const float4*)msum;

#pragma unroll 4
    for (int r = 0; r < 32; ++r) {
        int node = nBase + r;
        float4 v = hn4[(size_t)node * 32 + lane];
        ushort4 pv; pv.x = f2bf(v.x); pv.y = f2bf(v.y); pv.z = f2bf(v.z); pv.w = f2bf(v.w);
        *(ushort4*)&At[r][lane * 4] = pv;
        float c = cnt[node];
        float inv = 1.0f / ((c < 1.0f) ? 1.0f : c);
        float4 u = ms4[(size_t)node * 32 + lane];
        ushort4 pu;
        pu.x = f2bf(u.x * inv); pu.y = f2bf(u.y * inv);
        pu.z = f2bf(u.z * inv); pu.w = f2bf(u.w * inv);
        *(ushort4*)&At[r][128 + lane * 4] = pu;
    }
    __syncthreads();

    int m = lane & 15, hf = lane >> 4, col = m;

    v8f acc[2][8];
#pragma unroll
    for (int g = 0; g < 2; ++g)
#pragma unroll
        for (int nt = 0; nt < 8; ++nt)
#pragma unroll
            for (int i = 0; i < 8; ++i) acc[g][nt][i] = 0.0f;

#pragma unroll
    for (int kt = 0; kt < 8; ++kt) {
        v16bf a0 = load_afrag(&At[m][kt * 32], hf);
        v16bf a1 = load_afrag(&At[16 + m][kt * 32], hf);
#pragma unroll
        for (int nt = 0; nt < 8; ++nt) {
            v16bf bm = *(const v16bf*)(w1p + (((size_t)kt * 8 + nt) * 32 + lane) * 16);
            acc[0][nt] = __builtin_amdgcn_wmma_f32_16x16x32_bf16(
                false, a0, false, bm, (short)0, acc[0][nt], false, false);
            acc[1][nt] = __builtin_amdgcn_wmma_f32_16x16x32_bf16(
                false, a1, false, bm, (short)0, acc[1][nt], false, false);
        }
    }

#pragma unroll
    for (int nt = 0; nt < 8; ++nt) {
        float bias = b1[nt * 16 + col];
#pragma unroll
        for (int g = 0; g < 2; ++g)
#pragma unroll
            for (int r = 0; r < 8; ++r) {
                float x = acc[g][nt][r] + bias;
                Mt[g * 16 + r + 8 * hf][nt * 16 + col] = f2bf(silu_f(x));
            }
    }
    __syncthreads();

    v8f acc2[2][8];
#pragma unroll
    for (int g = 0; g < 2; ++g)
#pragma unroll
        for (int nt = 0; nt < 8; ++nt)
#pragma unroll
            for (int i = 0; i < 8; ++i) acc2[g][nt][i] = 0.0f;

#pragma unroll
    for (int kt = 0; kt < 4; ++kt) {
        v16bf a0 = load_afrag(&Mt[m][kt * 32], hf);
        v16bf a1 = load_afrag(&Mt[16 + m][kt * 32], hf);
#pragma unroll
        for (int nt = 0; nt < 8; ++nt) {
            v16bf bm = *(const v16bf*)(w2p + (((size_t)kt * 8 + nt) * 32 + lane) * 16);
            acc2[0][nt] = __builtin_amdgcn_wmma_f32_16x16x32_bf16(
                false, a0, false, bm, (short)0, acc2[0][nt], false, false);
            acc2[1][nt] = __builtin_amdgcn_wmma_f32_16x16x32_bf16(
                false, a1, false, bm, (short)0, acc2[1][nt], false, false);
        }
    }

#pragma unroll
    for (int nt = 0; nt < 8; ++nt) {
        float bias = b2[nt * 16 + col];
#pragma unroll
        for (int g = 0; g < 2; ++g)
#pragma unroll
            for (int r = 0; r < 8; ++r) {
                float s = silu_f(acc2[g][nt][r] + bias);
                int node = nBase + g * 16 + r + 8 * hf;
                size_t gi = (size_t)node * DM + nt * 16 + col;
                h[gi] = h[gi] + s;   // residual, unique (row,col) per lane
            }
    }
}

// ---------------- readout ----------------
__global__ void batch_sum_kernel(const float* __restrict__ h, const int* __restrict__ batch,
                                 float* __restrict__ bsum, float* __restrict__ bcnt) {
    int i = blockIdx.x, d = threadIdx.x;
    int b = batch[i];
    atomicAdd(&bsum[(size_t)b * DM + d], h[(size_t)i * DM + d]);
    if (d == 0) atomicAdd(&bcnt[b], 1.0f);
}

__global__ void ro_kernel(const float* __restrict__ bsum, const float* __restrict__ bcnt,
                          const float* __restrict__ w, const float* __restrict__ bias,
                          float* __restrict__ zcell) {
    int b = blockIdx.x, t = threadIdx.x;  // t in [0,256)
    float c = bcnt[b]; c = (c < 1.0f) ? 1.0f : c;
    float inv = 1.0f / c;
    float acc = 0.0f;
    for (int k = 0; k < DM; ++k)
        acc += (bsum[(size_t)b * DM + k] * inv) * w[(size_t)k * 256 + t];
    zcell[(size_t)b * 256 + t] = acc + bias[t];
}

__global__ void mlp1_kernel(const float* __restrict__ zcell, const float* __restrict__ w,
                            const float* __restrict__ bias, float* __restrict__ z) {
    int b = blockIdx.x, t = threadIdx.x;  // 256
    float acc = 0.0f;
    for (int k = 0; k < 256; ++k)
        acc += zcell[(size_t)b * 256 + k] * w[(size_t)k * 256 + t];
    z[(size_t)b * 256 + t] = gelu_f(acc + bias[t]);
}

__global__ void mlp2_kernel(const float* __restrict__ z, const float* __restrict__ w,
                            const float* __restrict__ bias, float* __restrict__ zproj) {
    int b = blockIdx.x, t = threadIdx.x;  // 128
    float acc = 0.0f;
    for (int k = 0; k < 256; ++k)
        acc += z[(size_t)b * 256 + k] * w[(size_t)k * DM + t];
    zproj[(size_t)b * DM + t] = acc + bias[t];
}

// ---------------- launch ----------------
extern "C" void kernel_launch(void* const* d_in, const int* in_sizes, int n_in,
                              void* d_out, int out_size, void* d_ws, size_t ws_size,
                              hipStream_t stream) {
    const int*   atom  = (const int*)d_in[0];
    const int*   eidx  = (const int*)d_in[1];     // [2,E]
    const float* rbf   = (const float*)d_in[2];
    const float* gram  = (const float*)d_in[3];
    const int*   batch = (const int*)d_in[4];
    const float* emb   = (const float*)d_in[5];
    const float* ln_g  = (const float*)d_in[6];
    const float* ln_b  = (const float*)d_in[7];
    const float* ew1   = (const float*)d_in[8];
    const float* eb1   = (const float*)d_in[9];
    const float* ew2   = (const float*)d_in[10];
    const float* eb2   = (const float*)d_in[11];
    const float* nw1   = (const float*)d_in[12];
    const float* nb1   = (const float*)d_in[13];
    const float* nw2   = (const float*)d_in[14];
    const float* nb2   = (const float*)d_in[15];
    const float* ro_w  = (const float*)d_in[16];
    const float* ro_b  = (const float*)d_in[17];
    const float* p1_w  = (const float*)d_in[18];
    const float* p1_b  = (const float*)d_in[19];
    const float* p2_w  = (const float*)d_in[20];
    const float* p2_b  = (const float*)d_in[21];

    char* wsp = (char*)d_ws;
    auto carve = [&](size_t bytes) -> char* {
        char* p = wsp;
        wsp += (bytes + 255) & ~(size_t)255;
        return p;
    };
    float* h    = (float*)carve((size_t)NN * DM * 4);
    float* hn   = (float*)carve((size_t)NN * DM * 4);
    float* msum = (float*)carve((size_t)NN * DM * 4);
    float* cnt  = (float*)carve((size_t)NN * 4);
    float* bsum = (float*)carve((size_t)BB * DM * 4);
    float* bcnt = (float*)carve((size_t)BB * 4);
    float* zbuf = (float*)carve((size_t)BB * 256 * 4);
    unsigned short* ew1p = (unsigned short*)carve((size_t)LNUM * EW1P_L * 2);
    unsigned short* ew2p = (unsigned short*)carve((size_t)LNUM * EW2P_L * 2);
    unsigned short* nw1p = (unsigned short*)carve((size_t)LNUM * NW1P_L * 2);
    unsigned short* nw2p = (unsigned short*)carve((size_t)LNUM * NW2P_L * 2);

    const int* srcIdx = eidx;
    const int* dstIdx = eidx + EE;

    // pack weights (bf16, WMMA B-fragment order)
    {
        int t1 = LNUM * 10 * 8 * 32;
        pack_w_kernel<<<(t1 + 255) / 256, 256, 0, stream>>>(ew1, ew1p, EIN, EINP, LNUM);
        int t2 = LNUM * 4 * 8 * 32;
        pack_w_kernel<<<(t2 + 255) / 256, 256, 0, stream>>>(ew2, ew2p, DM, DM, LNUM);
        int t3 = LNUM * 8 * 8 * 32;
        pack_w_kernel<<<(t3 + 255) / 256, 256, 0, stream>>>(nw1, nw1p, 256, 256, LNUM);
        pack_w_kernel<<<(t2 + 255) / 256, 256, 0, stream>>>(nw2, nw2p, DM, DM, LNUM);
    }

    embed_kernel<<<NN, DM, 0, stream>>>(atom, emb, h);

    zero_kernel<<<(NN + 255) / 256, 256, 0, stream>>>(cnt, (size_t)NN);
    degree_kernel<<<(EE + 255) / 256, 256, 0, stream>>>(dstIdx, cnt);

    for (int l = 0; l < LNUM; ++l) {
        ln_kernel<<<NN / 8, 256, 0, stream>>>(h, hn, ln_g + (size_t)l * DM, ln_b + (size_t)l * DM);
        zero_kernel<<<((size_t)NN * DM + 255) / 256, 256, 0, stream>>>(msum, (size_t)NN * DM);
        edge_kernel<<<EE / 32, 32, 0, stream>>>(
            hn, srcIdx, dstIdx, gram, rbf,
            ew1p + (size_t)l * EW1P_L, eb1 + (size_t)l * DM,
            ew2p + (size_t)l * EW2P_L, eb2 + (size_t)l * DM, msum);
        node_kernel<<<NN / 32, 32, 0, stream>>>(
            hn, msum, cnt,
            nw1p + (size_t)l * NW1P_L, nb1 + (size_t)l * DM,
            nw2p + (size_t)l * NW2P_L, nb2 + (size_t)l * DM, h);
    }

    zero_kernel<<<((size_t)BB * DM + 255) / 256, 256, 0, stream>>>(bsum, (size_t)BB * DM);
    zero_kernel<<<1, 256, 0, stream>>>(bcnt, (size_t)BB);
    batch_sum_kernel<<<NN, DM, 0, stream>>>(h, batch, bsum, bcnt);

    float* zcell_out = (float*)d_out;                     // (64,256)
    float* zproj_out = (float*)d_out + (size_t)BB * 256;  // (64,128)
    ro_kernel<<<BB, 256, 0, stream>>>(bsum, bcnt, ro_w, ro_b, zcell_out);
    mlp1_kernel<<<BB, 256, 0, stream>>>(zcell_out, p1_w, p1_b, zbuf);
    mlp2_kernel<<<BB, DM, 0, stream>>>(zbuf, p2_w, p2_b, zproj_out);
}
